// Segmenter_65721589563708
// MI455X (gfx1250) — matrix operand order repaired and checked by reading, not compiled, not run on hardware
//
#include <hip/hip_runtime.h>

// ---------------------------------------------------------------------------
// NeuralEF loss on MI455X (gfx1250, wave32).
// Dominant work: 8192x8192x768 cosine-similarity gram (103 GFLOP) via
// v_wmma_f32_16x16x32_bf16, fused with per-row top-32 selection.
// v3: 64 rows/block, B chunk staged in LDS (4x less L2 traffic), filled with
// GLOBAL_LOAD_ASYNC_TO_LDS_B128 (int4 AS1 -> int4 AS3 signature) overlapped
// with the top-k phase.
// ---------------------------------------------------------------------------

typedef __bf16 bf16_t;
typedef __attribute__((ext_vector_type(16))) __bf16 v16bf;
typedef __attribute__((ext_vector_type(8)))  __bf16 v8bf;
typedef __attribute__((ext_vector_type(8)))  float  v8f;
typedef __attribute__((ext_vector_type(4)))  int    v4i;

#define M_TOK   8192
#define D_FEAT  768
#define K_FEAT  32
#define KDIM    64
#define BS      8
#define HW      1024
#define SQRT_T  3.16227766016838f   // sqrt(10)
#define PIX_W   0.05f

#ifndef __has_builtin
#define __has_builtin(x) 0
#endif
#if __has_builtin(__builtin_amdgcn_global_load_async_to_lds_b128)
#define HAVE_ASYNC_LDS 1
#else
#define HAVE_ASYNC_LDS 0
#endif

typedef __attribute__((address_space(1))) v4i glob_v4i;   // global int4
typedef __attribute__((address_space(3))) v4i lds_v4i;    // LDS int4

__device__ __forceinline__ glob_v4i* to_glob16(const void* p) {
  return (glob_v4i*)(unsigned long long)p;
}
__device__ __forceinline__ lds_v4i* to_lds16(void* p) {
  // generic LDS address: low 32 bits are the LDS offset (ISA 10.2)
  return (lds_v4i*)(unsigned int)(unsigned long long)p;
}
__device__ __forceinline__ void wait_async_all() {
#if HAVE_ASYNC_LDS
#if __has_builtin(__builtin_amdgcn_s_wait_asynccnt)
  __builtin_amdgcn_s_wait_asynccnt(0);
#else
  asm volatile("s_wait_asynccnt 0" ::: "memory");
#endif
#endif
}

// ---------------------------------------------------------------------------
// K0: row-normalize features, convert to bf16.  One block per row.
// ---------------------------------------------------------------------------
__global__ __launch_bounds__(256) void normalize_kernel(
    const float* __restrict__ X, bf16_t* __restrict__ Xn) {
  __shared__ float red[256];
  const int row = blockIdx.x;
  const int t = threadIdx.x;
  const float* xr = X + (size_t)row * D_FEAT;
  float ss = 0.f;
  for (int k = t; k < D_FEAT; k += 256) { float v = xr[k]; ss += v * v; }
  red[t] = ss;
  __syncthreads();
  for (int s = 128; s > 0; s >>= 1) {
    if (t < s) red[t] += red[t + s];
    __syncthreads();
  }
  const float rn = rsqrtf(red[0]);
  for (int k = t; k < D_FEAT; k += 256)
    Xn[(size_t)row * D_FEAT + k] = (bf16_t)(xr[k] * rn);
}

// ---------------------------------------------------------------------------
// K1: pixel kNN graphs (two distance weightings), union, symmetric, bitmask.
// ---------------------------------------------------------------------------
__global__ __launch_bounds__(256) void pixel_knn_kernel(
    const float* __restrict__ im, unsigned* __restrict__ adj) {
  __shared__ float feat[HW][5];
  __shared__ float sq[HW];
  const int b = blockIdx.x;
  const int t = threadIdx.x;
  const float dws[2] = {2.0f, 0.1f};
  for (int wi = 0; wi < 2; ++wi) {
    const float dw = dws[wi];
    for (int p = t; p < HW; p += 256) {
      const int y = p >> 5, x = p & 31;
      const float r  = (im[((size_t)b * 3 + 0) * HW + p] + 1.f) * 0.5f;
      const float g  = (im[((size_t)b * 3 + 1) * HW + p] + 1.f) * 0.5f;
      const float bl = (im[((size_t)b * 3 + 2) * HW + p] + 1.f) * 0.5f;
      const float xf = ((float)x / 31.f) * dw;
      const float yf = ((float)y / 31.f) * dw;
      feat[p][0] = r;  feat[p][1] = g;  feat[p][2] = bl;
      feat[p][3] = xf; feat[p][4] = yf;
      sq[p] = r * r + g * g + bl * bl + xf * xf + yf * yf;
    }
    __syncthreads();
    for (int i = t; i < HW; i += 256) {
      float bv[10]; int bi[10];
#pragma unroll
      for (int e = 0; e < 10; ++e) { bv[e] = -1e30f; bi[e] = 0; }
      const float f0 = feat[i][0], f1 = feat[i][1], f2 = feat[i][2];
      const float f3 = feat[i][3], f4 = feat[i][4];
      for (int j = 0; j < HW; ++j) {
        if (j == i) continue;
        const float dot = f0 * feat[j][0] + f1 * feat[j][1] + f2 * feat[j][2] +
                          f3 * feat[j][3] + f4 * feat[j][4];
        const float d2 = sq[i] + sq[j] - 2.f * dot;
        const float e = -sqrtf(fmaxf(d2, 0.f));
        int mp = 0; float mv = bv[0];
#pragma unroll
        for (int q = 1; q < 10; ++q) if (bv[q] < mv) { mv = bv[q]; mp = q; }
        if (e > mv) { bv[mp] = e; bi[mp] = j; }
      }
      for (int q = 0; q < 10; ++q) {
        const int j = bi[q];
        atomicOr(&adj[((size_t)b * HW + i) * 32 + (j >> 5)], 1u << (j & 31));
        atomicOr(&adj[((size_t)b * HW + j) * 32 + (i >> 5)], 1u << (i & 31));
      }
    }
    __syncthreads();
  }
}

// ---------------------------------------------------------------------------
// K2: fused cosine-similarity (bf16 WMMA) + per-row running top-32.
// Block = 128 threads (4 waves), 64 rows (one 16-row strip per wave).
// A fragments (16x768 per wave) live in 192 VGPRs; B is staged per 64-col
// chunk in LDS (shared by all 4 waves -> 4x L2-traffic reduction) using
// async global->LDS copies overlapped with the top-k phase.
// ---------------------------------------------------------------------------
__global__ __launch_bounds__(128) void sim_topk_kernel(
    const bf16_t* __restrict__ Xn,
    float* __restrict__ topv, int* __restrict__ topi) {
  __shared__ bf16_t s_b[64][776];            // 64 cols x 768 bf16 (+8 pad)
  __shared__ float  s_tile[64][64];
  __shared__ float  s_lv[64][2][32];
  __shared__ unsigned short s_li[64][2][32];

  const int t = threadIdx.x;
  const int wave = t >> 5;
  const int lane = t & 31;
  const int h = lane >> 4;        // lane half (K-group select)
  const int ln16 = lane & 15;     // M for A-frag, N for B-frag
  const int r0 = blockIdx.x * 64;

  // ---- preload this wave's A fragments into registers (16 rows x 768) ----
  v16bf afr[24];
  {
    const bf16_t* arow = Xn + (size_t)(r0 + wave * 16 + ln16) * D_FEAT;
#pragma unroll
    for (int kk = 0; kk < 24; ++kk) {
      union { v16bf v; v8bf p[2]; } u;
      u.p[0] = *(const v8bf*)(arow + kk * 32 + h * 8);        // K kk*32+h*8 ..+7
      u.p[1] = *(const v8bf*)(arow + kk * 32 + 16 + h * 8);   // K +16
      afr[kk] = u.v;
    }
  }
  // ---- init per-thread top-32 sub-lists (2 per row) ----
  {
    const int row = t >> 1, sub = t & 1;
#pragma unroll 4
    for (int e = 0; e < 32; ++e) { s_lv[row][sub][e] = -1.0f; s_li[row][sub][e] = 0; }
  }

  // copy plan: thread t fills half a column row: 768B = 48 x 16B
  const int crow = t >> 1;
  const int chalf = t & 1;
  auto issue_copy = [&](int c0) {
    const size_t srcByte = ((size_t)(c0 + crow) * D_FEAT + (size_t)chalf * 384) * 2;
    char* dstGen = (char*)&s_b[crow][0] + chalf * 768;
#if HAVE_ASYNC_LDS
    glob_v4i* gp = to_glob16((const char*)Xn + srcByte);
    lds_v4i*  lp = to_lds16(dstGen);
#pragma unroll
    for (int i = 0; i < 48; ++i)
      __builtin_amdgcn_global_load_async_to_lds_b128(gp + i, lp + i, 0, 0);
#else
    const uint4* gp = (const uint4*)((const char*)Xn + srcByte);
    uint4* lp = (uint4*)dstGen;
#pragma unroll 8
    for (int i = 0; i < 48; ++i) lp[i] = gp[i];
#endif
  };

  issue_copy(0);
  float listMin = -1.0f;

  for (int c0 = 0; c0 < M_TOK; c0 += 64) {
    wait_async_all();
    __syncthreads();                       // B chunk visible to all waves

    // ---- compute: this wave's 16-row strip x 4 col tiles ----
    for (int ct = 0; ct < 4; ++ct) {
      const int c = ct * 16 + ln16;
      v8f acc = {};
#pragma unroll
      for (int kk = 0; kk < 24; ++kk) {
        union { v16bf v; v8bf p[2]; } bu;    // B: K h*8..+7 and 16+h*8..+7
        bu.p[0] = *(const v8bf*)(&s_b[c][kk * 32 + h * 8]);
        bu.p[1] = *(const v8bf*)(&s_b[c][kk * 32 + 16 + h * 8]);
        acc = __builtin_amdgcn_wmma_f32_16x16x32_bf16(
            false, afr[kk], false, bu.v, (short)0, acc, false, false);
      }
#pragma unroll
      for (int r = 0; r < 8; ++r) {
        const int m = r + h * 8;
        const int gr = r0 + wave * 16 + m;
        const int gc = c0 + ct * 16 + ln16;
        float v = acc[r];
        v = v > 0.f ? v : 0.f;
        if (gr == gc) v = -1.0f;             // kill diagonal
        s_tile[wave * 16 + m][ct * 16 + ln16] = v;
      }
    }
    __syncthreads();                       // s_tile ready; s_b fully consumed

    if (c0 + 64 < M_TOK) issue_copy(c0 + 64);   // overlap DMA with top-k

    // ---- top-k update: 2 threads per row, 32 cols each ----
    {
      const int row = t >> 1, sub = t & 1;
      float* lv = s_lv[row][sub];
      unsigned short* li = s_li[row][sub];
      for (int q = 0; q < 32; ++q) {
        const int cc = sub * 32 + q;
        const float v = s_tile[row][cc];
        if (v > listMin) {
          int mp = 0; float mv = lv[0];
#pragma unroll
          for (int e = 1; e < 32; ++e) if (lv[e] < mv) { mv = lv[e]; mp = e; }
          lv[mp] = v; li[mp] = (unsigned short)(c0 + cc);
          mv = lv[0];
#pragma unroll
          for (int e = 1; e < 32; ++e) mv = fminf(mv, lv[e]);
          listMin = mv;
        }
      }
    }
    __syncthreads();
  }

  // ---- merge the 2 sub-lists -> final top-32 per row ----
  if (t < 64) {
    const int row = t;
    for (int s = 0; s < K_FEAT; ++s) {
      float best = -2.0f; int bsub = 0, bpos = 0;
      for (int u2 = 0; u2 < 2; ++u2)
#pragma unroll
        for (int e = 0; e < 32; ++e) {
          const float v = s_lv[row][u2][e];
          if (v > best) { best = v; bsub = u2; bpos = e; }
        }
      topv[(size_t)(r0 + row) * K_FEAT + s] = best > 0.f ? best : 0.f;
      topi[(size_t)(r0 + row) * K_FEAT + s] = (int)s_li[row][bsub][bpos];
      s_lv[row][bsub][bpos] = -2.0f;
    }
  }
}

// ---------------------------------------------------------------------------
// K3: feature-graph degrees via symmetric scatter:  A_sym = (res + res^T)/2
// ---------------------------------------------------------------------------
__global__ void deg_feat_kernel(const float* __restrict__ topv,
                                const int* __restrict__ topi,
                                float* __restrict__ deg1) {
  const int id = blockIdx.x * blockDim.x + threadIdx.x;
  if (id >= M_TOK * K_FEAT) return;
  const int i = id >> 5;
  float v = topv[id]; if (v < 0.f) v = 0.f;
  const int j = topi[id];
  atomicAdd(&deg1[i], 0.5f * v);
  atomicAdd(&deg1[j], 0.5f * v);
}

// K4: degrees -> rsqrt (both graphs); pixel degree via popcount
__global__ void dinv_kernel(float* __restrict__ deg1, float* __restrict__ deg2,
                            const unsigned* __restrict__ adj) {
  const int i = blockIdx.x * blockDim.x + threadIdx.x;
  if (i >= M_TOK) return;
  int cnt = 0;
  for (int w = 0; w < 32; ++w) cnt += __popc(adj[(size_t)i * 32 + w]);
  deg2[i] = rsqrtf((float)cnt);
  deg1[i] = rsqrtf(deg1[i]);
}

// ---------------------------------------------------------------------------
// K5: sparse feature-gram * Psi  (symmetric scatter, atomics into gP)
// ---------------------------------------------------------------------------
__global__ __launch_bounds__(256) void gp_feat_kernel(
    const float* __restrict__ topv, const int* __restrict__ topi,
    const float* __restrict__ Dinv1, const float* __restrict__ Psi,
    float* __restrict__ gP) {
  const int i = blockIdx.x;
  const int t = threadIdx.x;
  const int lanek = t & 63;
  const int eg = t >> 6;
  const float di = Dinv1[i];
  for (int e = eg; e < K_FEAT; e += 4) {
    const float v = topv[(size_t)i * K_FEAT + e];
    if (v > 0.f) {
      const int j = topi[(size_t)i * K_FEAT + e];
      const float w = 0.5f * v * di * Dinv1[j];
      const float pj = Psi[(size_t)j * KDIM + lanek] * SQRT_T;
      const float pi = Psi[(size_t)i * KDIM + lanek] * SQRT_T;
      atomicAdd(&gP[(size_t)i * KDIM + lanek], w * pj);
      atomicAdd(&gP[(size_t)j * KDIM + lanek], w * pi);
    }
  }
}

// K6: sparse pixel-gram * Psi.  One block (64 threads) per node.
__global__ __launch_bounds__(64) void gp_pix_kernel(
    const unsigned* __restrict__ adj, const float* __restrict__ Dinv2,
    const float* __restrict__ Psi, float* __restrict__ gP) {
  __shared__ unsigned w_s[32];
  const int g = blockIdx.x;
  const int b = g >> 10;
  const int t = threadIdx.x;
  if (t < 32) w_s[t] = adj[(size_t)g * 32 + t];
  __syncthreads();
  const float dg = Dinv2[g];
  float acc = 0.f;
  for (int j = 0; j < HW; ++j) {
    if (w_s[j >> 5] & (1u << (j & 31))) {
      const int gj = (b << 10) + j;
      const float w = PIX_W * dg * Dinv2[gj];
      acc += w * Psi[(size_t)gj * KDIM + t] * SQRT_T;
    }
  }
  atomicAdd(&gP[(size_t)g * KDIM + t], acc);
}

// K7: R = Psi_s^T @ gP  (64x64, K=8192), split-K with atomics.
__global__ __launch_bounds__(256) void r_kernel(
    const float* __restrict__ Psi, const float* __restrict__ gP,
    float* __restrict__ R) {
  const int t = threadIdx.x;
  float acc[16];
#pragma unroll
  for (int q = 0; q < 16; ++q) acc[q] = 0.f;
  const int i0 = blockIdx.x * 64;
  for (int ii = 0; ii < 64; ++ii) {
    const int i = i0 + ii;
#pragma unroll
    for (int q = 0; q < 16; ++q) {
      const int idx = t * 16 + q;
      const int m = idx >> 6, n = idx & 63;
      acc[q] += Psi[(size_t)i * KDIM + m] * SQRT_T * gP[(size_t)i * KDIM + n];
    }
  }
#pragma unroll
  for (int q = 0; q < 16; ++q) atomicAdd(&R[t * 16 + q], acc[q]);
}

// K8: loss = -trace(R)/kdim ; reg = sum(triu(R^2,1))/kdim
__global__ __launch_bounds__(64) void final_kernel(
    const float* __restrict__ R, float* __restrict__ out) {
  __shared__ float s_tr[64], s_rg[64];
  const int t = threadIdx.x;
  float rg = 0.f;
  for (int n = t + 1; n < 64; ++n) { const float v = R[t * 64 + n]; rg += v * v; }
  s_tr[t] = R[t * 64 + t];
  s_rg[t] = rg;
  __syncthreads();
  for (int s = 32; s > 0; s >>= 1) {
    if (t < s) { s_tr[t] += s_tr[t + s]; s_rg[t] += s_rg[t + s]; }
    __syncthreads();
  }
  if (t == 0) { out[0] = -s_tr[0] / 64.f; out[1] = s_rg[0] / 64.f; }
}

// ---------------------------------------------------------------------------
extern "C" void kernel_launch(void* const* d_in, const int* in_sizes, int n_in,
                              void* d_out, int out_size, void* d_ws, size_t ws_size,
                              hipStream_t stream) {
  const float* hf  = (const float*)d_in[0];   // (8,1024,768)
  const float* Psi = (const float*)d_in[1];   // (8,1024,64)
  const float* im  = (const float*)d_in[2];   // (8,3,32,32)
  float* out = (float*)d_out;                 // (loss, reg)

  char* ws = (char*)d_ws;
  size_t o = 0;
  auto alloc = [&](size_t bytes) { void* p = ws + o; o = (o + bytes + 255) & ~(size_t)255; return p; };
  bf16_t*   Xn   = (bf16_t*)  alloc((size_t)M_TOK * D_FEAT * 2);
  float*    topv = (float*)   alloc((size_t)M_TOK * K_FEAT * 4);
  int*      topi = (int*)     alloc((size_t)M_TOK * K_FEAT * 4);
  unsigned* adj  = (unsigned*)alloc((size_t)BS * HW * 32 * 4);
  float*    deg1 = (float*)   alloc((size_t)M_TOK * 4);
  float*    deg2 = (float*)   alloc((size_t)M_TOK * 4);
  float*    gP   = (float*)   alloc((size_t)M_TOK * KDIM * 4);
  float*    R    = (float*)   alloc((size_t)KDIM * KDIM * 4);

  (void)hipMemsetAsync(adj,  0, (size_t)BS * HW * 32 * 4, stream);
  (void)hipMemsetAsync(deg1, 0, (size_t)M_TOK * 4, stream);
  (void)hipMemsetAsync(gP,   0, (size_t)M_TOK * KDIM * 4, stream);
  (void)hipMemsetAsync(R,    0, (size_t)KDIM * KDIM * 4, stream);

  normalize_kernel<<<M_TOK, 256, 0, stream>>>(hf, Xn);
  pixel_knn_kernel<<<BS, 256, 0, stream>>>(im, adj);
  sim_topk_kernel<<<M_TOK / 64, 128, 0, stream>>>(Xn, topv, topi);
  deg_feat_kernel<<<(M_TOK * K_FEAT + 255) / 256, 256, 0, stream>>>(topv, topi, deg1);
  dinv_kernel<<<M_TOK / 256, 256, 0, stream>>>(deg1, deg2, adj);
  gp_feat_kernel<<<M_TOK, 256, 0, stream>>>(topv, topi, deg1, Psi, gP);
  gp_pix_kernel<<<M_TOK, 64, 0, stream>>>(adj, deg2, Psi, gP);
  r_kernel<<<128, 256, 0, stream>>>(Psi, gP, R);
  final_kernel<<<1, 64, 0, stream>>>(R, out);
}
